// HungarianMatcher_52948356825308
// MI455X (gfx1250) — matrix operand order repaired and checked by reading, not compiled, not run on hardware
//
#include <hip/hip_runtime.h>
#include <hip/hip_bf16.h>

// Problem constants (fixed by the reference setup_inputs()).
#define Bn 4
#define Qn 300
#define NCn 80
#define Tn 100
#define Hn 256
#define Wn 256
#define Pn 12544          // divisible by 128 (12544 = 98*128)
#define BK 128            // K-slab per LDS stage

typedef __attribute__((ext_vector_type(16))) _Float16 v16h;
typedef __attribute__((ext_vector_type(8)))  float    v8f;

// ---------------------------------------------------------------------------
// Bilinear sample, align_corners=False, zero padding — exact reference math.
// ---------------------------------------------------------------------------
__device__ __forceinline__ float bilin(const float* __restrict__ m, float x, float y) {
    float ix = x * (float)Wn - 0.5f;
    float iy = y * (float)Hn - 0.5f;
    float x0 = floorf(ix), y0 = floorf(iy);
    float dx = ix - x0,    dy = iy - y0;

    auto g = [&](float xc, float yc) -> float {
        int xi = (int)fminf(fmaxf(xc, 0.0f), (float)(Wn - 1));
        int yi = (int)fminf(fmaxf(yc, 0.0f), (float)(Hn - 1));
        float valid = (xc >= 0.0f && xc <= (float)(Wn - 1) &&
                       yc >= 0.0f && yc <= (float)(Hn - 1)) ? 1.0f : 0.0f;
        return m[yi * Wn + xi] * valid;
    };
    float v00 = g(x0,        y0);
    float v01 = g(x0 + 1.0f, y0);
    float v10 = g(x0,        y0 + 1.0f);
    float v11 = g(x0 + 1.0f, y0 + 1.0f);
    return v00 * (1.0f - dx) * (1.0f - dy) + v01 * dx * (1.0f - dy)
         + v10 * (1.0f - dx) * dy          + v11 * dx * dy;
}

__device__ __forceinline__ float softplusf(float x) {
    // stable log(1 + e^x)
    return fmaxf(x, 0.0f) + log1pf(expf(-fabsf(x)));
}
__device__ __forceinline__ float sigmoidf_(float x) {
    return 1.0f / (1.0f + expf(-x));
}

// ---------------------------------------------------------------------------
// Kernel 1: sample target masks -> f16 [B,T,P], plus per-row sum (f32).
// One block per (b,t); deterministic in-block tree reduction.
// ---------------------------------------------------------------------------
__global__ void sample_tgt_kernel(const float* __restrict__ tgt_masks,
                                  const float* __restrict__ pts,
                                  _Float16* __restrict__ tgtS,
                                  float* __restrict__ tsum) {
    const int bt = blockIdx.x;                 // b*Tn + t
    const int b  = bt / Tn;
    const float* mask = tgt_masks + (size_t)bt * Hn * Wn;
    const float* pc   = pts + (size_t)b * Pn * 2;
    _Float16* dst     = tgtS + (size_t)bt * Pn;

    float s = 0.0f;
    for (int p = threadIdx.x; p < Pn; p += 256) {
        float v = bilin(mask, pc[2 * p], pc[2 * p + 1]);
        dst[p] = (_Float16)v;
        s += v;
    }
    __shared__ float red[256];
    red[threadIdx.x] = s;
    __syncthreads();
    for (int o = 128; o > 0; o >>= 1) {
        if (threadIdx.x < o) red[threadIdx.x] += red[threadIdx.x + o];
        __syncthreads();
    }
    if (threadIdx.x == 0) tsum[bt] = red[0];
}

// ---------------------------------------------------------------------------
// Kernel 2: sample pred masks -> raw f16 + sigmoid f16, plus f32 row sums of
// softplus(out) and sigmoid(out). One block per (b,q).
// ---------------------------------------------------------------------------
__global__ void sample_pred_kernel(const float* __restrict__ pred_masks,
                                   const float* __restrict__ pts,
                                   _Float16* __restrict__ outRaw,
                                   _Float16* __restrict__ outSig,
                                   float* __restrict__ ssum,
                                   float* __restrict__ sigsum) {
    const int bq = blockIdx.x;                 // b*Qn + q
    const int b  = bq / Qn;
    const float* mask = pred_masks + (size_t)bq * Hn * Wn;
    const float* pc   = pts + (size_t)b * Pn * 2;
    _Float16* dr = outRaw + (size_t)bq * Pn;
    _Float16* ds = outSig + (size_t)bq * Pn;

    float sp = 0.0f, sg = 0.0f;
    for (int p = threadIdx.x; p < Pn; p += 256) {
        float v = bilin(mask, pc[2 * p], pc[2 * p + 1]);
        float s = sigmoidf_(v);
        dr[p] = (_Float16)v;
        ds[p] = (_Float16)s;
        sp += softplusf(v);
        sg += s;
    }
    __shared__ float r0[256];
    __shared__ float r1[256];
    r0[threadIdx.x] = sp;
    r1[threadIdx.x] = sg;
    __syncthreads();
    for (int o = 128; o > 0; o >>= 1) {
        if (threadIdx.x < o) {
            r0[threadIdx.x] += r0[threadIdx.x + o];
            r1[threadIdx.x] += r1[threadIdx.x + o];
        }
        __syncthreads();
    }
    if (threadIdx.x == 0) { ssum[bq] = r0[0]; sigsum[bq] = r1[0]; }
}

// ---------------------------------------------------------------------------
// Kernel 3: focal classification-cost table ccls[b,q,c] = pos_cls - neg_cls.
// ---------------------------------------------------------------------------
__global__ void class_cost_kernel(const float* __restrict__ logits,
                                  float* __restrict__ ccls) {
    int idx = blockIdx.x * blockDim.x + threadIdx.x;
    if (idx >= Bn * Qn * NCn) return;
    float l = logits[idx];
    float p = sigmoidf_(l);
    float pos = 0.25f * (1.0f - p) * (1.0f - p) * (-logf(p + 1e-8f));         // ALPHA=0.25, GAMMA=2
    float neg = 0.75f * p * p * (-logf(1.0f - p + 1e-8f));
    ccls[idx] = pos - neg;
}

// ---------------------------------------------------------------------------
// Kernel 4: the two GEMMs (raw·tgtᵀ and sig·tgtᵀ) via v_wmma_f32_16x16x32_f16
// plus fused cost epilogue. 4 waves / workgroup; macro-tile 64(Q) x 16(T);
// K staged in LDS in 128-element slabs.
// ---------------------------------------------------------------------------
__global__ void __launch_bounds__(128)
cost_gemm_kernel(const _Float16* __restrict__ outRaw,
                 const _Float16* __restrict__ outSig,
                 const _Float16* __restrict__ tgtS,
                 const float* __restrict__ ssum,
                 const float* __restrict__ sigsum,
                 const float* __restrict__ tsum,
                 const float* __restrict__ ccls,
                 const int* __restrict__ labels,
                 float* __restrict__ out) {
    __shared__ _Float16 Araw[64][BK];   // 16 KB
    __shared__ _Float16 Asig[64][BK];   // 16 KB
    __shared__ _Float16 Btile[16][BK];  //  4 KB

    const int tid   = threadIdx.x;
    const int wave  = tid >> 5;
    const int lane  = tid & 31;
    const int qBase = blockIdx.x * 64;
    const int tBase = blockIdx.y * 16;
    const int b     = blockIdx.z;

    v8f accRaw = {};
    v8f accSig = {};

    union Frag { v16h v; uint4 u[2]; };

    const int mrow = lane & 15;
    const int hi   = lane >> 4;   // half-select per 16-bit WMMA operand layouts

    for (int ks = 0; ks < Pn; ks += BK) {
        // ---- cooperative stage: A panels (64 rows x 256B each) ----
        {
            const int r  = tid >> 1;               // 0..63
            const int c8 = (tid & 1) * 8;          // uint4 chunk start (16 per row)
            const int qr = min(qBase + r, Qn - 1);
            const size_t eoff = (size_t)(b * Qn + qr) * Pn + ks;
            const uint4* sr = (const uint4*)(outRaw + eoff);
            const uint4* ss = (const uint4*)(outSig + eoff);
            uint4* dr = (uint4*)&Araw[r][0];
            uint4* dsg = (uint4*)&Asig[r][0];
#pragma unroll
            for (int i = 0; i < 8; ++i) {
                dr[c8 + i]  = sr[c8 + i];
                dsg[c8 + i] = ss[c8 + i];
            }
        }
        // ---- cooperative stage: B panel (16 rows x 256B) ----
        {
            const int idx = tid * 2;               // 256 uint4 total / 128 thr
            const int br  = idx >> 4;
            const int bc  = idx & 15;
            const int tr  = min(tBase + br, Tn - 1);
            const uint4* bs = (const uint4*)(tgtS + ((size_t)(b * Tn + tr) * Pn + ks));
            uint4* bd = (uint4*)&Btile[br][0];
            bd[bc]     = bs[bc];
            bd[bc + 1] = bs[bc + 1];
        }
        __syncthreads();

        const uint4* arow = (const uint4*)&Araw[wave * 16 + mrow][0];  // 16 uint4 / row
        const uint4* srow = (const uint4*)&Asig[wave * 16 + mrow][0];
        const uint4* brow = (const uint4*)&Btile[mrow][0];
#pragma unroll
        for (int kk = 0; kk < BK / 32; ++kk) {
            // A frag (16-bit 16x32 layout): K = kk*32 + {hi*8..hi*8+7, 16+hi*8..}
            Frag fa, fs, fb;
            fa.u[0] = arow[kk * 4 + hi];
            fa.u[1] = arow[kk * 4 + hi + 2];
            fs.u[0] = srow[kk * 4 + hi];
            fs.u[1] = srow[kk * 4 + hi + 2];
            // B frag (32x16): K = kk*32 + {hi*8..hi*8+7, 16+hi*8..}
            fb.u[0] = brow[kk * 4 + hi];
            fb.u[1] = brow[kk * 4 + hi + 2];

            accRaw = __builtin_amdgcn_wmma_f32_16x16x32_f16(
                false, fa.v, false, fb.v, (short)0, accRaw, false, false);
            accSig = __builtin_amdgcn_wmma_f32_16x16x32_f16(
                false, fs.v, false, fb.v, (short)0, accSig, false, false);
        }
        __syncthreads();
    }

    // ---- epilogue: C layout — VGPR i holds M = i + 8*(lane>=16), N = lane&15
    const int t = tBase + mrow;
    float tS = 0.0f, cClBase = 0.0f;
    int lab = 0;
    if (t < Tn) {
        tS  = tsum[b * Tn + t];
        lab = labels[b * Tn + t];
    }
    const float invP = 1.0f / (float)Pn;
#pragma unroll
    for (int i = 0; i < 8; ++i) {
        const int q = qBase + wave * 16 + i + hi * 8;
        if (q < Qn && t < Tn) {
            const int bq = b * Qn + q;
            float cmask = (ssum[bq] - accRaw[i]) * invP;
            float cdice = 1.0f - (2.0f * accSig[i] + 1.0f) / (sigsum[bq] + tS + 1.0f);
            float ccl   = ccls[(size_t)bq * NCn + lab];
            out[(size_t)bq * Tn + t] =
                -(5.0f * cmask + 2.0f * ccl + 5.0f * cdice);
        }
    }
    (void)cClBase;
}

// ---------------------------------------------------------------------------
extern "C" void kernel_launch(void* const* d_in, const int* in_sizes, int n_in,
                              void* d_out, int out_size, void* d_ws, size_t ws_size,
                              hipStream_t stream) {
    const float* pred_logits = (const float*)d_in[0];   // [B,Q,NC]
    const float* pred_masks  = (const float*)d_in[1];   // [B,Q,H,W]
    const int*   tgt_labels  = (const int*)d_in[2];     // [B,T]
    const float* tgt_masks   = (const float*)d_in[3];   // [B,T,H,W]
    const float* point_coord = (const float*)d_in[4];   // [B,P,2]
    float* out = (float*)d_out;                         // [B,Q,T]

    // Workspace carve-up (~70.6 MB; all 16B-aligned by construction).
    const size_t nQP = (size_t)Bn * Qn * Pn;            // 15,052,800 f16
    const size_t nTP = (size_t)Bn * Tn * Pn;            //  5,017,600 f16
    _Float16* outRaw = (_Float16*)d_ws;
    _Float16* outSig = outRaw + nQP;
    _Float16* tgtS   = outSig + nQP;
    float* ssum   = (float*)(tgtS + nTP);
    float* sigsum = ssum + Bn * Qn;
    float* tsum   = sigsum + Bn * Qn;
    float* ccls   = tsum + Bn * Tn;

    sample_tgt_kernel<<<dim3(Bn * Tn), 256, 0, stream>>>(
        tgt_masks, point_coord, tgtS, tsum);
    sample_pred_kernel<<<dim3(Bn * Qn), 256, 0, stream>>>(
        pred_masks, point_coord, outRaw, outSig, ssum, sigsum);
    class_cost_kernel<<<dim3((Bn * Qn * NCn + 255) / 256), 256, 0, stream>>>(
        pred_logits, ccls);
    cost_gemm_kernel<<<dim3((Qn + 63) / 64, (Tn + 15) / 16, Bn), 128, 0, stream>>>(
        outRaw, outSig, tgtS, ssum, sigsum, tsum, ccls, tgt_labels, out);
}
